// LuongAttnDecoderRNN_15264313770348
// MI455X (gfx1250) — matrix operand order repaired and checked by reading, not compiled, not run on hardware
//
#include <hip/hip_runtime.h>
#include <hip/hip_bf16.h>
#include <cstddef>

// Shapes from the reference
#define BB 64
#define SS 512
#define HH 1024
#define VV 50257

typedef __attribute__((ext_vector_type(16))) __bf16 v16bf;
typedef __attribute__((ext_vector_type(8)))  float  v8f;

__device__ __forceinline__ float sigmoidf_(float x) { return 1.0f / (1.0f + __expf(-x)); }

// ---------------------------------------------------------------------------
// WMMA fragment loaders (wave32 layouts per CDNA5 ISA 7.12.2)
// A (16x32, 16-bit): lanes 0-15 -> M=lane, K in {0..7,16..23}; lanes 16-31 -> M=lane-16, K in {8..15,24..31}
// B (32x16, 16-bit): N = lane%16; lanes 0-15 hold K in {0..7,16..23}, lanes 16-31 hold K in {8..15,24..31} (2 per VGPR)
// C/D (16x16 f32):   VGPR r -> M = r + 8*(lane>=16); N = lane%16
// ---------------------------------------------------------------------------
__device__ __forceinline__ v16bf frag_a_bf16(const __bf16* Arow0, int lda, int k0, int lane) {
  int m  = lane & 15;
  int kb = (lane & 16) ? 8 : 0;
  const __bf16* p = Arow0 + (size_t)m * lda + k0 + kb;
  v16bf a;
#pragma unroll
  for (int j = 0; j < 4; ++j) {
    a[2 * j]     = p[2 * j];
    a[2 * j + 1] = p[2 * j + 1];
    a[8 + 2 * j] = p[16 + 2 * j];
    a[9 + 2 * j] = p[16 + 2 * j + 1];
  }
  return a;
}

__device__ __forceinline__ v16bf frag_a_f32(const float* Arow0, int lda, int k0, int lane) {
  int m  = lane & 15;
  int kb = (lane & 16) ? 8 : 0;
  const float* p = Arow0 + (size_t)m * lda + k0 + kb;
  v16bf a;
#pragma unroll
  for (int j = 0; j < 4; ++j) {
    a[2 * j]     = (__bf16)p[2 * j];
    a[2 * j + 1] = (__bf16)p[2 * j + 1];
    a[8 + 2 * j] = (__bf16)p[16 + 2 * j];
    a[9 + 2 * j] = (__bf16)p[16 + 2 * j + 1];
  }
  return a;
}

// B^T fragment from row-major W (N x K), fp32 -> bf16 on the fly; N-edge clamped.
__device__ __forceinline__ v16bf frag_b_wt(const float* W, int ldw, int n0, int k0, int N, int lane) {
  int n = n0 + (lane & 15);
  if (n >= N) n = N - 1;
  int kb = (lane & 16) ? 8 : 0;
  const float* p = W + (size_t)n * ldw + k0 + kb;
  v16bf b;
#pragma unroll
  for (int j = 0; j < 8; ++j) {
    b[j]     = (__bf16)p[j];
    b[8 + j] = (__bf16)p[16 + j];
  }
  return b;
}

// ---------------------------------------------------------------------------
// Generic dual-operand WMMA GEMM:  C = act( A0 * W0^T + A1 * W1^T + bias0 + bias1 )
// A*: (M,K) bf16 row-major. W*: (N,K) fp32 row-major with row stride ldw*.
// One 16x16 tile per wave; K loop of x32 bf16 WMMA steps.
// ---------------------------------------------------------------------------
__global__ void wmma_gemm(const __bf16* __restrict__ A0, const float* __restrict__ W0, int ldw0,
                          const __bf16* __restrict__ A1, const float* __restrict__ W1, int ldw1,
                          const float* __restrict__ bias0, const float* __restrict__ bias1,
                          float* __restrict__ C, __bf16* __restrict__ Cbf,
                          int M, int N, int K, int act) {
  const int lane = threadIdx.x & 31;
  const int wave = threadIdx.x >> 5;
  const int tiles_m = M >> 4;
  const int tiles_n = (N + 15) >> 4;
  const int tile = blockIdx.x * (blockDim.x >> 5) + wave;
  if (tile >= tiles_m * tiles_n) return;  // wave-uniform -> EXEC stays all-ones for WMMA
  const int mt = tile % tiles_m;
  const int nt = tile / tiles_m;
  const int row0 = mt << 4;
  const int n0 = nt << 4;

  v8f acc = {0.f, 0.f, 0.f, 0.f, 0.f, 0.f, 0.f, 0.f};
  for (int k0 = 0; k0 < K; k0 += 32) {
    // hint the streaming weight path (lowers to global_prefetch_b8)
    __builtin_prefetch(W0 + (size_t)(n0 + (lane & 15)) * ldw0 + k0 + 64, 0, 1);
    v16bf a = frag_a_bf16(A0 + (size_t)row0 * K, K, k0, lane);
    v16bf b = frag_b_wt(W0, ldw0, n0, k0, N, lane);
    acc = __builtin_amdgcn_wmma_f32_16x16x32_bf16(false, a, false, b, (short)0, acc, false, false);
    if (A1) {
      v16bf a1 = frag_a_bf16(A1 + (size_t)row0 * K, K, k0, lane);
      v16bf b1 = frag_b_wt(W1, ldw1, n0, k0, N, lane);
      acc = __builtin_amdgcn_wmma_f32_16x16x32_bf16(false, a1, false, b1, (short)0, acc, false, false);
    }
  }

  const int col = n0 + (lane & 15);
  if (col < N) {
    const int mbase = (lane & 16) ? 8 : 0;
#pragma unroll
    for (int i = 0; i < 8; ++i) {
      const int row = row0 + mbase + i;
      float v = acc[i];
      if (bias0) v += bias0[col];
      if (bias1) v += bias1[col];
      if (act)   v = tanhf(v);
      if (C)   C[(size_t)row * N + col] = v;
      if (Cbf) Cbf[(size_t)row * N + col] = (__bf16)v;
    }
  }
}

// ---------------------------------------------------------------------------
// Fused attention scores:
//   scores[s*B+b] = sum_g tanh( (enc @ We^T)[s*B+b, g] + hWh[b, g] ) * v[g]
// enc: (S,B,H) fp32 flat -> rows of the (S*B, H) GEMM. One 16-row tile per wave,
// N grouped by 4 tiles to amortize A-fragment loads; epilogue fused; LDS ds_add
// reduction across lanes (16 columns each) into per-row partial sums.
// ---------------------------------------------------------------------------
__global__ void scores_wmma(const float* __restrict__ enc, const float* __restrict__ We, int ldwe,
                            const float* __restrict__ hWh, const float* __restrict__ vat,
                            float* __restrict__ scores) {
  __shared__ float ssc[8][16];
  const int lane = threadIdx.x & 31;
  const int wave = threadIdx.x >> 5;
  const int rt = blockIdx.x * 8 + wave;   // 16-row tile of the (S*B) dimension
  const int row0 = rt << 4;
  if (lane < 16) ssc[wave][lane] = 0.f;

  float part[8];
#pragma unroll
  for (int i = 0; i < 8; ++i) part[i] = 0.f;

  const int n = lane & 15;
  const int mbase = (lane & 16) ? 8 : 0;
  const v8f zero = {0.f, 0.f, 0.f, 0.f, 0.f, 0.f, 0.f, 0.f};

  for (int gt = 0; gt < (HH >> 4); gt += 4) {
    v8f acc[4];
#pragma unroll
    for (int j = 0; j < 4; ++j) acc[j] = zero;

    for (int k0 = 0; k0 < HH; k0 += 32) {
      v16bf a = frag_a_f32(enc + (size_t)row0 * HH, HH, k0, lane);
#pragma unroll
      for (int j = 0; j < 4; ++j) {
        v16bf b = frag_b_wt(We, ldwe, (gt + j) << 4, k0, HH, lane);
        acc[j] = __builtin_amdgcn_wmma_f32_16x16x32_bf16(false, a, false, b, (short)0, acc[j], false, false);
      }
    }
    // fused epilogue: energy = tanh(encWe + hWh); partial dot with v_attn
#pragma unroll
    for (int j = 0; j < 4; ++j) {
      const int g = ((gt + j) << 4) + n;
      const float vg = vat[g];
#pragma unroll
      for (int i = 0; i < 8; ++i) {
        const int row = row0 + mbase + i;     // row = s*B + b
        const int bb = row & (BB - 1);        // B = 64
        part[i] += tanhf(acc[j][i] + hWh[(size_t)bb * HH + g]) * vg;
      }
    }
  }
#pragma unroll
  for (int i = 0; i < 8; ++i) atomicAdd(&ssc[wave][mbase + i], part[i]);  // ds_add_f32
  __syncthreads();
  if (lane < 16) scores[row0 + lane] = ssc[wave][lane];  // (S,B) layout
}

// ---------------------------------------------------------------------------
// Elementwise / small kernels
// ---------------------------------------------------------------------------
__global__ void prep_kernel(const int* __restrict__ seq, const float* __restrict__ emb,
                            const float* __restrict__ h0,
                            __bf16* __restrict__ x_bf, __bf16* __restrict__ h_bf) {
  const int idx = blockIdx.x * blockDim.x + threadIdx.x;  // B*H
  const int b = idx >> 10, h = idx & (HH - 1);
  x_bf[idx] = (__bf16)emb[(size_t)seq[b] * HH + h];
  h_bf[idx] = (__bf16)h0[idx];
}

__global__ void lstm_elem(const float* __restrict__ gates, const float* __restrict__ c0,
                          float* __restrict__ h_out, float* __restrict__ c_out,
                          __bf16* __restrict__ h_bf) {
  const int idx = blockIdx.x * blockDim.x + threadIdx.x;  // B*H
  const int b = idx >> 10, h = idx & (HH - 1);
  const float* g = gates + (size_t)b * 4 * HH;
  const float ig = sigmoidf_(g[h]);
  const float fg = sigmoidf_(g[HH + h]);
  const float gg = tanhf(g[2 * HH + h]);
  const float og = sigmoidf_(g[3 * HH + h]);
  const float c = fg * c0[idx] + ig * gg;
  const float hn = og * tanhf(c);
  c_out[idx] = c;
  h_out[idx] = hn;
  h_bf[idx] = (__bf16)hn;
}

// softmax over batch axis (axis=0 of (B,S)) per s, then mask multiply. 64 threads/block.
__global__ void softmax_over_b(const float* __restrict__ scores, const float* __restrict__ mask,
                               float* __restrict__ a_sm) {
  __shared__ float red[BB];
  const int s = blockIdx.x;
  const int b = threadIdx.x;
  const float v = scores[(size_t)s * BB + b];
  red[b] = v; __syncthreads();
  for (int off = BB / 2; off > 0; off >>= 1) {
    if (b < off) red[b] = fmaxf(red[b], red[b + off]);
    __syncthreads();
  }
  const float mx = red[0]; __syncthreads();
  const float e = __expf(v - mx);
  red[b] = e; __syncthreads();
  for (int off = BB / 2; off > 0; off >>= 1) {
    if (b < off) red[b] += red[b + off];
    __syncthreads();
  }
  a_sm[(size_t)s * BB + b] = (e / red[0]) * mask[(size_t)s * BB + b];
}

// per-b renormalization over s; writes alpha in (B,S) layout (== alpha output (B,1,S))
__global__ void renorm_alpha(const float* __restrict__ a_sm, float* __restrict__ alpha) {
  __shared__ float red[256];
  const int b = blockIdx.x;
  const int tid = threadIdx.x;
  float acc = 0.f;
  for (int s = tid; s < SS; s += 256) acc += a_sm[(size_t)s * BB + b];
  red[tid] = acc; __syncthreads();
  for (int off = 128; off > 0; off >>= 1) {
    if (tid < off) red[tid] += red[tid + off];
    __syncthreads();
  }
  const float inv = 1.0f / (red[0] + 1e-10f);
  for (int s = tid; s < SS; s += 256) alpha[(size_t)b * SS + s] = a_sm[(size_t)s * BB + b] * inv;
}

// context[b,h] = sum_s alpha[b,s] * enc[s,b,h]   (enc L2-resident: 134MB vs 192MB L2)
__global__ void context_kernel(const float* __restrict__ alpha, const float* __restrict__ enc,
                               float* __restrict__ ctx, __bf16* __restrict__ ctx_bf) {
  const int b = blockIdx.y;
  const int h = blockIdx.x * blockDim.x + threadIdx.x;
  float acc = 0.f;
  for (int s = 0; s < SS; ++s)
    acc += alpha[(size_t)b * SS + s] * enc[((size_t)s * BB + b) * HH + h];
  ctx[(size_t)b * HH + h] = acc;
  ctx_bf[(size_t)b * HH + h] = (__bf16)acc;
}

// ---------------------------------------------------------------------------
// Launcher
// ---------------------------------------------------------------------------
extern "C" void kernel_launch(void* const* d_in, const int* in_sizes, int n_in,
                              void* d_out, int out_size, void* d_ws, size_t ws_size,
                              hipStream_t stream) {
  const int*   input_seq = (const int*)  d_in[0];
  const float* h0        = (const float*)d_in[2];
  const float* c0        = (const float*)d_in[3];
  const float* enc       = (const float*)d_in[4];   // (S,B,H)
  const float* inp_mask  = (const float*)d_in[5];   // (S,B)
  const float* emb       = (const float*)d_in[6];
  const float* W_ih      = (const float*)d_in[7];   // (4H,H)
  const float* W_hh      = (const float*)d_in[8];
  const float* b_ih      = (const float*)d_in[9];
  const float* b_hh      = (const float*)d_in[10];
  const float* W_attn    = (const float*)d_in[11];  // (H,2H): [Wh | We]
  const float* v_attn    = (const float*)d_in[12];
  const float* W_concat  = (const float*)d_in[13];  // (H,2H): [Wc1 | Wc2]
  const float* b_concat  = (const float*)d_in[14];
  const float* W_out     = (const float*)d_in[15];  // (V,H)
  const float* b_out     = (const float*)d_in[16];

  // d_out: output(B,V) | context(B,H) | h(1,B,H) | c(1,B,H) | alpha(B,1,S)
  float* out_output  = (float*)d_out;
  float* out_context = out_output + (size_t)BB * VV;
  float* out_h       = out_context + (size_t)BB * HH;
  float* out_c       = out_h + (size_t)BB * HH;
  float* out_alpha   = out_c + (size_t)BB * HH;

  // workspace carve-out (256B aligned)
  char* w = (char*)d_ws;
  auto take = [&](size_t bytes) { char* p = w; w += (bytes + 255) & ~size_t(255); return p; };
  __bf16* x_bf    = (__bf16*)take((size_t)BB * HH * 2);
  __bf16* hprev_bf= (__bf16*)take((size_t)BB * HH * 2);
  float*  gates   = (float*) take((size_t)BB * 4 * HH * 4);
  __bf16* hnew_bf = (__bf16*)take((size_t)BB * HH * 2);
  float*  hWh     = (float*) take((size_t)BB * HH * 4);
  float*  scores  = (float*) take((size_t)SS * BB * 4);
  float*  a_sm    = (float*) take((size_t)SS * BB * 4);
  __bf16* ctx_bf  = (__bf16*)take((size_t)BB * HH * 2);
  __bf16* cat_bf  = (__bf16*)take((size_t)BB * HH * 2);

  const int thr = 256;
  auto gemm_blocks = [](int M, int N) {
    const int tiles = (M >> 4) * ((N + 15) >> 4);
    return (tiles + 7) / 8;  // 8 waves per 256-thread block
  };

  // 1. embedding gather + bf16 casts of x and h_prev
  prep_kernel<<<(BB * HH) / thr, thr, 0, stream>>>(input_seq, emb, h0, x_bf, hprev_bf);

  // 2. LSTM gates = x@W_ih^T + h@W_hh^T + b_ih + b_hh   (dual-operand WMMA)
  wmma_gemm<<<gemm_blocks(BB, 4 * HH), thr, 0, stream>>>(
      x_bf, W_ih, HH, hprev_bf, W_hh, HH, b_ih, b_hh,
      gates, nullptr, BB, 4 * HH, HH, 0);

  // 3. LSTM cell elementwise -> h_new, c_new (+ bf16 h_new)
  lstm_elem<<<(BB * HH) / thr, thr, 0, stream>>>(gates, c0, out_h, out_c, hnew_bf);

  // 4. hWh = h_new @ Wh^T   (Wh = W_attn[:, :H], row stride 2H)
  wmma_gemm<<<gemm_blocks(BB, HH), thr, 0, stream>>>(
      hnew_bf, W_attn, 2 * HH, nullptr, nullptr, 0, nullptr, nullptr,
      hWh, nullptr, BB, HH, HH, 0);

  // 5. fused scores: tanh(enc@We^T + hWh) . v_attn    (We = W_attn[:, H:])
  scores_wmma<<<(SS * BB / 16) / 8, thr, 0, stream>>>(
      enc, W_attn + HH, 2 * HH, hWh, v_attn, scores);

  // 6. softmax over batch axis, mask; 7. renormalize over s -> alpha
  softmax_over_b<<<SS, BB, 0, stream>>>(scores, inp_mask, a_sm);
  renorm_alpha<<<BB, thr, 0, stream>>>(a_sm, out_alpha);

  // 8. context = sum_s alpha * enc
  context_kernel<<<dim3(HH / thr, BB), thr, 0, stream>>>(out_alpha, enc, out_context, ctx_bf);

  // 9. concat = tanh(h_new@Wc1^T + context@Wc2^T + b_concat)
  wmma_gemm<<<gemm_blocks(BB, HH), thr, 0, stream>>>(
      hnew_bf, W_concat, 2 * HH, ctx_bf, W_concat + HH, 2 * HH, b_concat, nullptr,
      nullptr, cat_bf, BB, HH, HH, 1);

  // 10. output = concat @ W_out^T + b_out  (V-edge guarded; streams 206MB of W_out once)
  wmma_gemm<<<gemm_blocks(BB, VV), thr, 0, stream>>>(
      cat_bf, W_out, HH, nullptr, nullptr, 0, b_out, nullptr,
      out_output, nullptr, BB, VV, HH, 0);

  (void)in_sizes; (void)n_in; (void)out_size; (void)ws_size;
}
